// MonodomainPINN_15272903704622
// MI455X (gfx1250) — matrix operand order repaired and checked by reading, not compiled, not run on hardware
//
#include <hip/hip_runtime.h>

// ---------------------------------------------------------------------------
// Monodomain PINN residual, fused DeepONet forward + forward-mode derivatives.
// N=65536 points, M=128, H=P=256.  All GEMMs via v_wmma_f32_16x16x32_bf16.
// ---------------------------------------------------------------------------

#define NPTS 65536
#define MB   128
#define HN   256

typedef __attribute__((ext_vector_type(16))) __bf16 v16bf;
typedef __attribute__((ext_vector_type(8)))  float  v8f;

struct U128 { unsigned int a, b, c, d; };
union  V16U { v16bf v; U128 q[2]; };

// Branch-free fast tanh: tanh(x) = 1 - 2/(exp(2x)+1).
// __expf -> v_exp_f32, rcp -> v_rcp_f32; no EXEC divergence in epilogues.
static __device__ __forceinline__ float fast_tanh(float x) {
  float xc = fminf(fmaxf(x, -15.0f), 15.0f);
  float e  = __expf(2.0f * xc);
  return 1.0f - 2.0f * __builtin_amdgcn_rcpf(e + 1.0f);
}

// ---- A operand: 16xK tile from LDS (bf16, row stride ld elements) ----------
// lanes 0-15: row m=lane, K {k0..k0+7, k0+16..k0+23}
// lanes16-31: row m=lane-16, K {k0+8..k0+15, k0+24..k0+31}
static __device__ __forceinline__ v16bf load_A_lds(const __bf16* s, int lane, int ld, int k0) {
  const int m  = lane & 15;
  const int kb = k0 + ((lane >> 4) ? 8 : 0);
  const __bf16* p = s + m * ld + kb;
  V16U u;
  u.q[0] = *(const U128*)(p);
  u.q[1] = *(const U128*)(p + 16);
  return u.v;
}

// ---- B operand: Kx16 tile from pre-transposed weights WT[n][k] -------------
// lanes 0-15: col n0+lane,    K {k0..k0+7, k0+16..k0+23}
// lanes16-31: col n0+lane-16, K {k0+8..k0+15, k0+24..k0+31}
static __device__ __forceinline__ v16bf load_B_glb(const __bf16* wt, int lane, int ldk, int n0, int k0) {
  const int c  = n0 + (lane & 15);
  const int kb = k0 + ((lane >> 4) ? 8 : 0);
  const __bf16* p = wt + c * ldk + kb;
  V16U u;
  u.q[0] = *(const U128*)(p);
  u.q[1] = *(const U128*)(p + 16);
  return u.v;
}

static __device__ __forceinline__ v8f wmma_bf16(v16bf a, v16bf b, v8f c) {
  return __builtin_amdgcn_wmma_f32_16x16x32_bf16(false, a, false, b, (short)0, c, false, false);
}

// C/D layout: VGPR r -> row (r + 8*(lane>=16)), col n0 + (lane&15)
static __device__ __forceinline__ void store_tile(__bf16* s, int lane, int ld, int n0, const float* v) {
  const int mb = (lane >> 4) * 8;
  const int n  = n0 + (lane & 15);
#pragma unroll
  for (int r = 0; r < 8; ++r) s[(mb + r) * ld + n] = (__bf16)v[r];
}

// Prefetch this wave's B-tile rows of the next weight matrix into cache
// (lowers to global_prefetch_b8; one cacheline per lane, covers k=0..63).
static __device__ __forceinline__ void prefetch_W(const __bf16* wt, int lane, int ldk, int n0) {
  __builtin_prefetch(wt + (n0 + (lane & 15) + ((lane >> 4) ? 16 : 0)) * ldk, 0, 1);
}

// ---------------------------------------------------------------------------
// weight fp32 (K x Nc, row major) -> bf16 transposed (Nc x K, row major)
// ---------------------------------------------------------------------------
__global__ void k_transpose_bf16(const float* __restrict__ src, __bf16* __restrict__ dst,
                                 int K, int Nc) {
  int i = blockIdx.x * blockDim.x + threadIdx.x;
  if (i >= K * Nc) return;
  int n = i / K, k = i - n * K;
  dst[n * K + k] = (__bf16)src[k * Nc + n];
}

// ---------------------------------------------------------------------------
// Fused kernel: one workgroup = 16 points, 8 waves, wave w owns columns
// [32w, 32w+32) of every 16x256 activation tile (2 WMMA N-tiles).
// ---------------------------------------------------------------------------
__global__ __launch_bounds__(256)
void k_pinn_fused(const float* __restrict__ branch_in, const float* __restrict__ trunk_in,
                  const __bf16* __restrict__ WbT1, const __bf16* __restrict__ WbT2,
                  const __bf16* __restrict__ WbT3, const __bf16* __restrict__ WbT4,
                  const __bf16* __restrict__ WtT2, const __bf16* __restrict__ WtT3,
                  const __bf16* __restrict__ WtT4,
                  const float* __restrict__ bb1, const float* __restrict__ bb2,
                  const float* __restrict__ bb3, const float* __restrict__ bb4,
                  const float* __restrict__ Wt1, const float* __restrict__ bt1,
                  const float* __restrict__ bt2, const float* __restrict__ bt3,
                  const float* __restrict__ bt4, const float* __restrict__ b0p,
                  float* __restrict__ out) {
  // streams: 0=v 1=dx 2=dy 3=dz 4=dxx 5=dyy (branch reuses slots 0/1)
  __shared__ __align__(16) __bf16 sS[6][16 * HN];   // 48 KB
  __shared__ __align__(16) __bf16 sBF[16 * HN];     //  8 KB branch features
  __shared__ float sDot[4][16];                     // u, u_t, u_xx, u_yy

  const int tid  = threadIdx.x;
  const int wave = tid >> 5;
  const int lane = tid & 31;
  const int row0 = blockIdx.x * 16;
  const int n0   = wave * 32;

  if (tid < 64) ((float*)sDot)[tid] = 0.0f;

  prefetch_W(WbT1, lane, MB, n0);

  // ---- stage branch input tile (16 x 128) fp32 -> bf16 in sS[1] ----
  for (int i = tid; i < 16 * MB; i += 256) {
    int m = i >> 7, k = i & (MB - 1);
    sS[1][m * MB + k] = (__bf16)branch_in[(size_t)(row0 + m) * MB + k];
  }
  __syncthreads();

  // ================= branch layer 1 (K=128, tanh) -> sS[0] =================
  {
    v8f a0 = {}, a1 = {};
#pragma unroll
    for (int k0 = 0; k0 < MB; k0 += 32) {
      v16bf A  = load_A_lds(sS[1], lane, MB, k0);
      v16bf Bt = load_B_glb(WbT1, lane, MB, n0, k0);
      v16bf Bu = load_B_glb(WbT1, lane, MB, n0 + 16, k0);
      a0 = wmma_bf16(A, Bt, a0);
      a1 = wmma_bf16(A, Bu, a1);
    }
    prefetch_W(WbT2, lane, HN, n0);
    float c0 = bb1[n0 + (lane & 15)], c1 = bb1[n0 + 16 + (lane & 15)];
    float v0[8], v1[8];
#pragma unroll
    for (int r = 0; r < 8; ++r) { v0[r] = fast_tanh(a0[r] + c0); v1[r] = fast_tanh(a1[r] + c1); }
    store_tile(sS[0], lane, HN, n0, v0);
    store_tile(sS[0], lane, HN, n0 + 16, v1);
  }
  __syncthreads();

  // ================= branch layers 2,3 (in-place sS[0], tanh) ==============
  const __bf16* WbH[3] = { WbT2, WbT3, WbT4 };
  const float*  bbH[2] = { bb2, bb3 };
  for (int L = 0; L < 2; ++L) {
    v8f a0 = {}, a1 = {};
#pragma unroll
    for (int k0 = 0; k0 < HN; k0 += 32) {
      v16bf A  = load_A_lds(sS[0], lane, HN, k0);
      v16bf Bt = load_B_glb(WbH[L], lane, HN, n0, k0);
      v16bf Bu = load_B_glb(WbH[L], lane, HN, n0 + 16, k0);
      a0 = wmma_bf16(A, Bt, a0);
      a1 = wmma_bf16(A, Bu, a1);
    }
    prefetch_W(WbH[L + 1], lane, HN, n0);
    __syncthreads();                       // all reads done before in-place write
    float c0 = bbH[L][n0 + (lane & 15)], c1 = bbH[L][n0 + 16 + (lane & 15)];
    float v0[8], v1[8];
#pragma unroll
    for (int r = 0; r < 8; ++r) { v0[r] = fast_tanh(a0[r] + c0); v1[r] = fast_tanh(a1[r] + c1); }
    store_tile(sS[0], lane, HN, n0, v0);
    store_tile(sS[0], lane, HN, n0 + 16, v1);
    __syncthreads();
  }

  // ================= branch layer 4 (linear) -> sBF ========================
  {
    v8f a0 = {}, a1 = {};
#pragma unroll
    for (int k0 = 0; k0 < HN; k0 += 32) {
      v16bf A  = load_A_lds(sS[0], lane, HN, k0);
      v16bf Bt = load_B_glb(WbT4, lane, HN, n0, k0);
      v16bf Bu = load_B_glb(WbT4, lane, HN, n0 + 16, k0);
      a0 = wmma_bf16(A, Bt, a0);
      a1 = wmma_bf16(A, Bu, a1);
    }
    prefetch_W(WtT2, lane, HN, n0);
    float c0 = bb4[n0 + (lane & 15)], c1 = bb4[n0 + 16 + (lane & 15)];
    float v0[8], v1[8];
#pragma unroll
    for (int r = 0; r < 8; ++r) { v0[r] = a0[r] + c0; v1[r] = a1[r] + c1; }
    store_tile(sBF, lane, HN, n0, v0);
    store_tile(sBF, lane, HN, n0 + 16, v1);
  }
  __syncthreads();   // branch reads of sS[0] done; trunk may now fill streams

  // ================= trunk layer 1 (analytic, K=3) -> 6 streams ============
  for (int i = tid; i < 16 * HN; i += 256) {
    int m = i >> 8, n = i & (HN - 1);
    int pt = row0 + m;
    float x = trunk_in[pt * 3 + 0], y = trunk_in[pt * 3 + 1], z = trunk_in[pt * 3 + 2];
    float w0 = Wt1[n], w1 = Wt1[HN + n], w2 = Wt1[2 * HN + n];
    float h  = fast_tanh(x * w0 + y * w1 + z * w2 + bt1[n]);
    float hp = 1.0f - h * h;
    sS[0][i] = (__bf16)h;
    sS[1][i] = (__bf16)(hp * w0);
    sS[2][i] = (__bf16)(hp * w1);
    sS[3][i] = (__bf16)(hp * w2);
    sS[4][i] = (__bf16)(-2.0f * h * hp * w0 * w0);   // a_xx = 0 at layer 1
    sS[5][i] = (__bf16)(-2.0f * h * hp * w1 * w1);
  }
  __syncthreads();

  // ========== trunk layers 2,3: 6 streams, B reused across streams =========
  const __bf16* WtH[3] = { WtT2, WtT3, WtT4 };
  const float*  btH[2] = { bt2, bt3 };
  for (int L = 0; L < 2; ++L) {
    v8f acc[6][2] = {};
#pragma unroll
    for (int k0 = 0; k0 < HN; k0 += 32) {
      v16bf Bt = load_B_glb(WtH[L], lane, HN, n0, k0);
      v16bf Bu = load_B_glb(WtH[L], lane, HN, n0 + 16, k0);
#pragma unroll
      for (int st = 0; st < 6; ++st) {
        v16bf A = load_A_lds(sS[st], lane, HN, k0);
        acc[st][0] = wmma_bf16(A, Bt, acc[st][0]);
        acc[st][1] = wmma_bf16(A, Bu, acc[st][1]);
      }
    }
    prefetch_W(WtH[L + 1], lane, HN, n0);
    __syncthreads();                        // all stream reads done (in-place)
#pragma unroll
    for (int t = 0; t < 2; ++t) {
      float bv = btH[L][n0 + 16 * t + (lane & 15)];
      float o[6][8];
#pragma unroll
      for (int r = 0; r < 8; ++r) {
        float h  = fast_tanh(acc[0][t][r] + bv);
        float hp = 1.0f - h * h;
        float ax = acc[1][t][r], ay = acc[2][t][r], az = acc[3][t][r];
        o[0][r] = h;
        o[1][r] = hp * ax;
        o[2][r] = hp * ay;
        o[3][r] = hp * az;
        o[4][r] = hp * acc[4][t][r] - 2.0f * h * hp * ax * ax;
        o[5][r] = hp * acc[5][t][r] - 2.0f * h * hp * ay * ay;
      }
#pragma unroll
      for (int st = 0; st < 6; ++st) store_tile(sS[st], lane, HN, n0 + 16 * t, o[st]);
    }
    __syncthreads();
  }

  // ===== trunk layer 4 (linear, streams v/z/xx/yy) fused with bf . t =======
  {
    const int slot[4] = { 0, 3, 4, 5 };
    v8f acc[4][2] = {};
#pragma unroll
    for (int k0 = 0; k0 < HN; k0 += 32) {
      v16bf Bt = load_B_glb(WtT4, lane, HN, n0, k0);
      v16bf Bu = load_B_glb(WtT4, lane, HN, n0 + 16, k0);
#pragma unroll
      for (int st = 0; st < 4; ++st) {
        v16bf A = load_A_lds(sS[slot[st]], lane, HN, k0);
        acc[st][0] = wmma_bf16(A, Bt, acc[st][0]);
        acc[st][1] = wmma_bf16(A, Bu, acc[st][1]);
      }
    }
    // partial dot over this wave's 32 columns, straight from accumulators
    float part[4][8];
#pragma unroll
    for (int st = 0; st < 4; ++st)
#pragma unroll
      for (int r = 0; r < 8; ++r) part[st][r] = 0.0f;
    const int mb = (lane >> 4) * 8;
#pragma unroll
    for (int t = 0; t < 2; ++t) {
      int n = n0 + 16 * t + (lane & 15);
      float b4 = bt4[n];
#pragma unroll
      for (int r = 0; r < 8; ++r) {
        float bfv = (float)sBF[(mb + r) * HN + n];
        part[0][r] += bfv * (acc[0][t][r] + b4);   // u  (bias inside dot)
        part[1][r] += bfv * acc[1][t][r];          // u_t
        part[2][r] += bfv * acc[2][t][r];          // u_xx
        part[3][r] += bfv * acc[3][t][r];          // u_yy
      }
    }
    // reduce over 16 columns (half-wave), then across waves via LDS atomics
#pragma unroll
    for (int st = 0; st < 4; ++st)
#pragma unroll
      for (int r = 0; r < 8; ++r) {
        float v = part[st][r];
        v += __shfl_xor(v, 1, 32);
        v += __shfl_xor(v, 2, 32);
        v += __shfl_xor(v, 4, 32);
        v += __shfl_xor(v, 8, 32);
        if ((lane & 15) == 0) atomicAdd(&sDot[st][mb + r], v);
      }
  }
  __syncthreads();

  // ================= residual: 16 threads, one per point ===================
  if (tid < 16) {
    int pt = row0 + tid;
    float u   = sDot[0][tid] + b0p[0];
    float ut  = sDot[1][tid];
    float uxx = sDot[2][tid];
    float uyy = sDot[3][tid];
    float x = trunk_in[pt * 3 + 0], y = trunk_in[pt * 3 + 1];
    float sig = 0.1f;                                 // SIGMA_H
    { float dx = x - 0.3f, dy = y - 0.7f; if (dx * dx + dy * dy < 0.01f)   sig = 0.01f; }
    { float dx = x - 0.7f, dy = y - 0.3f; if (dx * dx + dy * dy < 0.0225f) sig = 0.01f; }
    { float dx = x - 0.5f, dy = y - 0.5f; if (dx * dx + dy * dy < 0.01f)   sig = 0.01f; }
    float fu = u * (u - 0.1f) * (u - 1.0f);           // A*(u-FR)*(u-FT)*(u-FD)
    out[pt] = ut - sig * (uxx + uyy) + fu;
  }
}

// ---------------------------------------------------------------------------
extern "C" void kernel_launch(void* const* d_in, const int* in_sizes, int n_in,
                              void* d_out, int out_size, void* d_ws, size_t ws_size,
                              hipStream_t stream) {
  const float* branch_in = (const float*)d_in[0];
  const float* trunk_in  = (const float*)d_in[1];
  const float* Wb1 = (const float*)d_in[2];  const float* bb1 = (const float*)d_in[3];
  const float* Wt1 = (const float*)d_in[4];  const float* bt1 = (const float*)d_in[5];
  const float* Wb2 = (const float*)d_in[6];  const float* bb2 = (const float*)d_in[7];
  const float* Wt2 = (const float*)d_in[8];  const float* bt2 = (const float*)d_in[9];
  const float* Wb3 = (const float*)d_in[10]; const float* bb3 = (const float*)d_in[11];
  const float* Wt3 = (const float*)d_in[12]; const float* bt3 = (const float*)d_in[13];
  const float* Wb4 = (const float*)d_in[14]; const float* bb4 = (const float*)d_in[15];
  const float* Wt4 = (const float*)d_in[16]; const float* bt4 = (const float*)d_in[17];
  const float* b0p = (const float*)d_in[18];

  __bf16* ws   = (__bf16*)d_ws;
  __bf16* WbT1 = ws;                       // 256 x 128
  __bf16* WbT2 = WbT1 + HN * MB;           // 256 x 256
  __bf16* WbT3 = WbT2 + HN * HN;
  __bf16* WbT4 = WbT3 + HN * HN;
  __bf16* WtT2 = WbT4 + HN * HN;
  __bf16* WtT3 = WtT2 + HN * HN;
  __bf16* WtT4 = WtT3 + HN * HN;

  const int thr = 256;
  k_transpose_bf16<<<(MB * HN + thr - 1) / thr, thr, 0, stream>>>(Wb1, WbT1, MB, HN);
  k_transpose_bf16<<<(HN * HN + thr - 1) / thr, thr, 0, stream>>>(Wb2, WbT2, HN, HN);
  k_transpose_bf16<<<(HN * HN + thr - 1) / thr, thr, 0, stream>>>(Wb3, WbT3, HN, HN);
  k_transpose_bf16<<<(HN * HN + thr - 1) / thr, thr, 0, stream>>>(Wb4, WbT4, HN, HN);
  k_transpose_bf16<<<(HN * HN + thr - 1) / thr, thr, 0, stream>>>(Wt2, WtT2, HN, HN);
  k_transpose_bf16<<<(HN * HN + thr - 1) / thr, thr, 0, stream>>>(Wt3, WtT3, HN, HN);
  k_transpose_bf16<<<(HN * HN + thr - 1) / thr, thr, 0, stream>>>(Wt4, WtT4, HN, HN);

  k_pinn_fused<<<NPTS / 16, 256, 0, stream>>>(
      branch_in, trunk_in,
      WbT1, WbT2, WbT3, WbT4, WtT2, WtT3, WtT4,
      bb1, bb2, bb3, bb4,
      Wt1, bt1, bt2, bt3, bt4, b0p,
      (float*)d_out);
}
